// MultiHeadAttention_2422361555248
// MI455X (gfx1250) — matrix operand order repaired and checked by reading, not compiled, not run on hardware
//
#include <hip/hip_runtime.h>

#define BSZ   2
#define SEQ   2048
#define HDIM  1024
#define NHEAD 16
#define HEAD  64

typedef __attribute__((ext_vector_type(16))) _Float16 v16h;
typedef __attribute__((ext_vector_type(8)))  _Float16 v8h;
typedef __attribute__((ext_vector_type(8)))  float    v8f;

__device__ __forceinline__ float fast_exp2(float x) {
#if __has_builtin(__builtin_amdgcn_exp2f)
    return __builtin_amdgcn_exp2f(x);
#else
    return exp2f(x);
#endif
}

// ---- gfx1250 async global->LDS copy (ASYNCcnt) --------------------------------
__device__ __forceinline__ unsigned lds_addr_of(const void* p) {
    // shared->generic cast puts the LDS byte offset in addr[31:0]
    return (unsigned)(size_t)p;
}
__device__ __forceinline__ void async_copy_b128(unsigned lds, const _Float16* g) {
    asm volatile("global_load_async_to_lds_b128 %0, %1, off"
                 :: "v"(lds), "v"(g) : "memory");
}
#define WAIT_ASYNC(n) asm volatile("s_wait_asynccnt %0" :: "n"(n) : "memory")

// ---- WMMA fragment helpers ----------------------------------------------------
__device__ __forceinline__ v16h make_frag(v8h lo, v8h hi) {
    v16h r;
#pragma unroll
    for (int i = 0; i < 8; ++i) { r[i] = lo[i]; r[i + 8] = hi[i]; }
    return r;
}

// A fragment 16x32 f16 from row-major [M][K] (ld in halves).
__device__ __forceinline__ v16h load_a_frag(const _Float16* p, int ld) {
    const int lane = threadIdx.x & 31;
    const int row  = lane & 15;
    const int kb   = (lane >> 4) << 3;
    const _Float16* q = p + (size_t)row * ld + kb;
    return make_frag(*(const v8h*)(q), *(const v8h*)(q + 16));
}

// B fragment 32x16 f16 from row-major [K][N] (ld in halves): lane = N col, k in table order.
__device__ __forceinline__ v16h load_b_frag(const _Float16* p, int ld) {
    const int lane = threadIdx.x & 31;
    const int col  = lane & 15;
    const int kb   = (lane >> 4) << 3;
    v16h r;
#pragma unroll
    for (int i = 0; i < 8; ++i) {
        r[i]     = p[(size_t)(kb + i) * ld + col];
        r[i + 8] = p[(size_t)(kb + 16 + i) * ld + col];
    }
    return r;
}

__device__ __forceinline__ v8f wmma32(v16h a, v16h b, v8f c) {
    return __builtin_amdgcn_wmma_f32_16x16x32_f16(false, a, false, b, (short)0, c,
                                                  false, false);
}

// ---------------------------------------------------------------------------
// f32 -> f16 conversion (prepass so every GEMM operand is async-copyable)
// ---------------------------------------------------------------------------
__global__ __launch_bounds__(256)
void f32_to_f16_kernel(const float* __restrict__ s, _Float16* __restrict__ d, int n) {
    const int i = (blockIdx.x * 256 + threadIdx.x) * 8;
    if (i < n) {
#pragma unroll
        for (int j = 0; j < 8; ++j) d[i + j] = (_Float16)s[i + j];
    }
}

// ---------------------------------------------------------------------------
// C[M,N] = A[M,K]@W[K,N] + bias (all f16 in, f32 acc). 128x64 tile, 256 thr.
// Double-buffered LDS fed by async global->LDS copies (3 b128 per thread/tile).
// KT epilogue scatters K projection into Kt[b,h,feat,sk].
// ---------------------------------------------------------------------------
#define LDA_T 40
#define LDB_T 72

template <bool KT, typename TC>
__global__ __launch_bounds__(256)
void gemm_bias_wmma(const _Float16* __restrict__ A, const _Float16* __restrict__ W,
                    const float* __restrict__ bias, TC* __restrict__ C,
                    int N, int Kd) {
    __shared__ alignas(16) _Float16 ldsA[2][128 * LDA_T];
    __shared__ alignas(16) _Float16 ldsB[2][32 * LDB_T];

    const int n0   = blockIdx.x << 6;
    const int m0   = blockIdx.y << 7;
    const int tid  = threadIdx.x;
    const int wave = tid >> 5;
    const int lane = tid & 31;

    v8f acc[4];
#pragma unroll
    for (int j = 0; j < 4; ++j)
#pragma unroll
        for (int r = 0; r < 8; ++r) acc[j][r] = 0.f;

    // A tile: 128x32 halves = 512 x 16B chunks -> 2 per thread
    const int ar0 = (tid * 2) >> 2,       as0 = ((tid * 2) & 3) << 3;
    const int ar1 = (tid * 2 + 1) >> 2,   as1 = ((tid * 2 + 1) & 3) << 3;
    // W tile: 32x64 halves = 256 x 16B chunks -> 1 per thread
    const int br  = tid >> 3,             bs  = (tid & 7) << 3;

    auto stage = [&](int kt, int buf) {
        const int k0 = kt << 5;
        async_copy_b128(lds_addr_of(&ldsA[buf][ar0 * LDA_T + as0]),
                        A + (size_t)(m0 + ar0) * Kd + k0 + as0);
        async_copy_b128(lds_addr_of(&ldsA[buf][ar1 * LDA_T + as1]),
                        A + (size_t)(m0 + ar1) * Kd + k0 + as1);
        async_copy_b128(lds_addr_of(&ldsB[buf][br * LDB_T + bs]),
                        W + (size_t)(k0 + br) * N + n0 + bs);
    };

    const int T = Kd >> 5;
    stage(0, 0);
    for (int t = 0; t < T; ++t) {
        if (t + 1 < T) { stage(t + 1, (t + 1) & 1); WAIT_ASYNC(3); }
        else           { WAIT_ASYNC(0); }
        __syncthreads();                       // tile t fully in LDS for all waves
        const int buf = t & 1;
        v16h a = load_a_frag(&ldsA[buf][(wave << 4) * LDA_T], LDA_T);
#pragma unroll
        for (int j = 0; j < 4; ++j)
            acc[j] = wmma32(a, load_b_frag(&ldsB[buf][j << 4], LDB_T), acc[j]);
        __syncthreads();                       // done reading buf before reuse
    }

    const int rbase = m0 + (wave << 4) + ((lane >> 4) << 3);
    const int cl    = lane & 15;
#pragma unroll
    for (int j = 0; j < 4; ++j) {
        const int col = n0 + (j << 4) + cl;
        const float bv = bias[col];
#pragma unroll
        for (int r = 0; r < 8; ++r) {
            const float v = acc[j][r] + bv;
            if constexpr (KT) {
                // row=(b*SEQ+sk), col=(h*64+f) -> Kt[((b*16+h)*64+f)*SEQ + sk]
                const int row = rbase + r;
                const int bb = row >> 11, sk = row & (SEQ - 1);
                const int hh = col >> 6,  f  = col & 63;
                C[(((size_t)bb * NHEAD + hh) * HEAD + f) * SEQ + sk] = (TC)v;
            } else {
                C[(size_t)(rbase + r) * N + col] = (TC)v;
            }
        }
    }
}

// ---------------------------------------------------------------------------
// Attention: WG = one (b,h) x 64 q-rows; 4 independent waves (16 rows each).
// Score B-frags stream straight from global Kt[b,h,feat,sk] (L2-hot, no
// barriers, no transpose). Two-pass online softmax; attn written normalized.
// ---------------------------------------------------------------------------
__global__ __launch_bounds__(128)
void attn_wmma(const _Float16* __restrict__ Q, const _Float16* __restrict__ Kt,
               const _Float16* __restrict__ V, float* __restrict__ attn,
               _Float16* __restrict__ attended) {
    __shared__ alignas(16) _Float16 ldsP[4][16 * 32];   // per-wave P restage

    const int bh   = blockIdx.y;
    const int b    = bh >> 4;
    const int h    = bh & 15;
    const int q0   = blockIdx.x << 6;
    const int wave = threadIdx.x >> 5;
    const int lane = threadIdx.x & 31;
    const int mrow0 = q0 + (wave << 4);

    const _Float16* Qh  = Q  + (size_t)b * SEQ * HDIM + h * HEAD;
    const _Float16* Kth = Kt + (size_t)bh * HEAD * SEQ;          // [64][2048]
    const _Float16* Vh  = V  + (size_t)b * SEQ * HDIM + h * HEAD;
    float* attn_bh = attn + (size_t)bh * SEQ * SEQ;

    const v16h aq0 = load_a_frag(Qh + (size_t)mrow0 * HDIM, HDIM);       // feat 0..31
    const v16h aq1 = load_a_frag(Qh + (size_t)mrow0 * HDIM + 32, HDIM);  // feat 32..63

    const float scale = 0.03125f;  // 1/sqrt(1024) per reference
    const float L2E   = 1.4426950408889634f;

    float mrun[8], srun[8];
#pragma unroll
    for (int r = 0; r < 8; ++r) { mrun[r] = -3.0e38f; srun[r] = 0.f; }

    // ---------- pass 1: running row max / sum ----------
    for (int t = 0; t < SEQ / 64; ++t) {
        const int sk0 = t << 6;
        float s[4][8];
#pragma unroll
        for (int sub = 0; sub < 4; ++sub) {
            const int n0 = sk0 + (sub << 4);
            v8f c;
#pragma unroll
            for (int r = 0; r < 8; ++r) c[r] = 0.f;
            c = wmma32(aq0, load_b_frag(Kth + n0, SEQ), c);
            c = wmma32(aq1, load_b_frag(Kth + 32 * SEQ + n0, SEQ), c);
#pragma unroll
            for (int r = 0; r < 8; ++r) s[sub][r] = c[r] * scale;
        }
#pragma unroll
        for (int r = 0; r < 8; ++r) {
            float tm = fmaxf(fmaxf(s[0][r], s[1][r]), fmaxf(s[2][r], s[3][r]));
#pragma unroll
            for (int msk = 1; msk < 16; msk <<= 1) tm = fmaxf(tm, __shfl_xor(tm, msk, 32));
            const float nm = fmaxf(mrun[r], tm);
            float es = 0.f;
#pragma unroll
            for (int sub = 0; sub < 4; ++sub) es += fast_exp2((s[sub][r] - nm) * L2E);
#pragma unroll
            for (int msk = 1; msk < 16; msk <<= 1) es += __shfl_xor(es, msk, 32);
            srun[r] = srun[r] * fast_exp2((mrun[r] - nm) * L2E) + es;
            mrun[r] = nm;
        }
    }

    float inv[8];
#pragma unroll
    for (int r = 0; r < 8; ++r) inv[r] = 1.f / srun[r];

    v8f acc[4];
#pragma unroll
    for (int j = 0; j < 4; ++j)
#pragma unroll
        for (int r = 0; r < 8; ++r) acc[j][r] = 0.f;

    // ---------- pass 2: write normalized attn, attended += P @ V ----------
    _Float16* Pw = &ldsP[wave][0];
    for (int t = 0; t < SEQ / 32; ++t) {
        const int skb = t << 5;
        float pv[2][8];
#pragma unroll
        for (int sub = 0; sub < 2; ++sub) {
            const int n0 = skb + (sub << 4);
            v8f c;
#pragma unroll
            for (int r = 0; r < 8; ++r) c[r] = 0.f;
            c = wmma32(aq0, load_b_frag(Kth + n0, SEQ), c);
            c = wmma32(aq1, load_b_frag(Kth + 32 * SEQ + n0, SEQ), c);
#pragma unroll
            for (int r = 0; r < 8; ++r) {
                const float p = fast_exp2((c[r] * scale - mrun[r]) * L2E) * inv[r];
                pv[sub][r] = p;
                attn_bh[(size_t)(mrow0 + r + ((lane >> 4) << 3)) * SEQ +
                        n0 + (lane & 15)] = p;
            }
        }
        __builtin_amdgcn_wave_barrier();
#pragma unroll
        for (int sub = 0; sub < 2; ++sub)
#pragma unroll
            for (int r = 0; r < 8; ++r)
                Pw[(r + ((lane >> 4) << 3)) * 32 + (sub << 4) + (lane & 15)] =
                    (_Float16)pv[sub][r];
        __builtin_amdgcn_wave_barrier();
        const v16h ap = load_a_frag(Pw, 32);
#pragma unroll
        for (int j = 0; j < 4; ++j) {
            const v16h bv = load_b_frag(Vh + (size_t)skb * HDIM + (j << 4), HDIM);
            acc[j] = wmma32(ap, bv, acc[j]);
        }
    }

    const int rb = mrow0 + ((lane >> 4) << 3);
#pragma unroll
    for (int j = 0; j < 4; ++j)
#pragma unroll
        for (int r = 0; r < 8; ++r)
            attended[((size_t)b * SEQ + rb + r) * HDIM + h * HEAD + (j << 4) +
                     (lane & 15)] = (_Float16)acc[j][r];
}

// ---------------------------------------------------------------------------
extern "C" void kernel_launch(void* const* d_in, const int* in_sizes, int n_in,
                              void* d_out, int out_size, void* d_ws, size_t ws_size,
                              hipStream_t stream) {
    (void)in_sizes; (void)n_in; (void)out_size; (void)ws_size;
    const float* inputs  = (const float*)d_in[0];
    const float* context = (const float*)d_in[1];
    const float* Wq = (const float*)d_in[2]; const float* bq = (const float*)d_in[3];
    const float* Wk = (const float*)d_in[4]; const float* bk = (const float*)d_in[5];
    const float* Wv = (const float*)d_in[6]; const float* bv = (const float*)d_in[7];
    const float* Wo = (const float*)d_in[8]; const float* bo = (const float*)d_in[9];

    float* out  = (float*)d_out;
    float* attn = out + (size_t)BSZ * SEQ * HDIM;          // tuple order: (out, attn)

    const size_t actElems = (size_t)BSZ * SEQ * HDIM;      // 4M
    const size_t wElems   = (size_t)HDIM * HDIM;           // 1M
    _Float16* p   = (_Float16*)d_ws;
    _Float16* Xf  = p; p += actElems;      // inputs f16
    _Float16* Cf  = p; p += actElems;      // context f16
    _Float16* wq  = p; p += wElems;
    _Float16* wk  = p; p += wElems;
    _Float16* wv  = p; p += wElems;
    _Float16* wo  = p; p += wElems;
    _Float16* Qw  = p; p += actElems;
    _Float16* Ktw = p; p += actElems;      // K transposed per head: [b,h,feat,sk]
    _Float16* Vw  = p; p += actElems;
    _Float16* Att = p;                      // total ws use: 56 MB

    // f32 -> f16 prepasses
    f32_to_f16_kernel<<<(int)(actElems / 2048), 256, 0, stream>>>(inputs,  Xf, (int)actElems);
    f32_to_f16_kernel<<<(int)(actElems / 2048), 256, 0, stream>>>(context, Cf, (int)actElems);
    f32_to_f16_kernel<<<(int)(wElems / 2048),   256, 0, stream>>>(Wq, wq, (int)wElems);
    f32_to_f16_kernel<<<(int)(wElems / 2048),   256, 0, stream>>>(Wk, wk, (int)wElems);
    f32_to_f16_kernel<<<(int)(wElems / 2048),   256, 0, stream>>>(Wv, wv, (int)wElems);
    f32_to_f16_kernel<<<(int)(wElems / 2048),   256, 0, stream>>>(Wo, wo, (int)wElems);

    const dim3 gp(HDIM / 64, (BSZ * SEQ) / 128);           // (16, 32)
    gemm_bias_wmma<false, _Float16><<<gp, 256, 0, stream>>>(Xf, wq, bq, Qw,  HDIM, HDIM);
    gemm_bias_wmma<true,  _Float16><<<gp, 256, 0, stream>>>(Cf, wk, bk, Ktw, HDIM, HDIM);
    gemm_bias_wmma<false, _Float16><<<gp, 256, 0, stream>>>(Cf, wv, bv, Vw,  HDIM, HDIM);

    attn_wmma<<<dim3(SEQ / 64, BSZ * NHEAD), 128, 0, stream>>>(Qw, Ktw, Vw, attn, Att);

    gemm_bias_wmma<false, float><<<gp, 256, 0, stream>>>(Att, wo, bo, out, HDIM, HDIM);
}
